// Model_82291573392147
// MI455X (gfx1250) — matrix operand order repaired and checked by reading, not compiled, not run on hardware
//
#include <hip/hip_runtime.h>
#include <hip/hip_bf16.h>

typedef _Float16 h16;
typedef __attribute__((ext_vector_type(16))) _Float16 v16h;
typedef __attribute__((ext_vector_type(8)))  float    v8f;

#define BB 16
#define NB 500
#define QB 25
#define CB 80
#define RB 3
#define DB 512
#define EB 3072
#define NPAD 512                 /* padded node dim for adj GEMM */
#define MROWS (BB*NB)            /* 8000 */
#define MQ    (BB*QB)            /* 400 */

// ---------------------------------------------------------------------------
// Branchless f16 fragment loaders (layouts per CDNA5 ISA 7.12.2, wave32).
// A fragment 16x32: lane m=lane&15; halves at k = kb+0..7 and kb+16..23,
// kb = k0 + (lane>=16 ? 8 : 0)  -> two contiguous 16B loads.
// B fragment 32x16 from N-major ("transposed") storage Bt[n][k]:
// lane n=lane&15; halves at k = kb+0..7 and kb+16..23, kb = k0 + (lane>=16 ? 8 : 0)
// -> two contiguous 16B loads.
// ---------------------------------------------------------------------------
union fragU { v16h h; uint4 q[2]; };

__device__ __forceinline__ v16h load_a16(const h16* A, int lda, int m0, int k0, int lane) {
  int m = m0 + (lane & 15);
  int kb = k0 + ((lane >> 4) << 3);
  const h16* p = A + (size_t)m * lda + kb;
  fragU f;
  f.q[0] = *(const uint4*)p;
  f.q[1] = *(const uint4*)(p + 16);
  return f.h;
}

__device__ __forceinline__ v16h load_bt16(const h16* Bt, int ldk, int n0, int k0, int lane) {
  int n = n0 + (lane & 15);
  int kb = k0 + ((lane >> 4) << 3);
  const h16* p = Bt + (size_t)n * ldk + kb;
  fragU f;
  f.q[0] = *(const uint4*)p;
  f.q[1] = *(const uint4*)(p + 16);
  return f.h;
}

__device__ __forceinline__ v8f wmma32(v16h a, v16h b, v8f c) {
  return __builtin_amdgcn_wmma_f32_16x16x32_f16(false, a, false, b, (short)0, c, false, false);
}

// ---------------------------------------------------------------------------
// Elementwise prep kernels
// ---------------------------------------------------------------------------
__global__ void k_cvt_f16(const float* __restrict__ src, h16* __restrict__ dst, int n) {
  int i = blockIdx.x * blockDim.x + threadIdx.x;
  if (i < n) dst[i] = (h16)src[i];
}

// W (K x N, row-major f32) -> Wt (N x K, row-major f16)
__global__ void k_tcvt(const float* __restrict__ W, h16* __restrict__ Wt, int K, int N) {
  int i = blockIdx.x * blockDim.x + threadIdx.x;
  if (i >= K * N) return;
  int k = i / N, n = i - k * N;
  Wt[(size_t)n * K + k] = (h16)W[i];
}

// adj (B,R,500,500 f32) -> adjp (B,R,512,512 f16, zero padded)
__global__ void k_padadj(const float* __restrict__ adj, h16* __restrict__ adjp) {
  int i = blockIdx.x * blockDim.x + threadIdx.x;
  if (i >= BB * RB * NPAD * NPAD) return;
  int col = i & (NPAD - 1);
  int row = (i >> 9) & (NPAD - 1);
  int br = i >> 18;
  h16 v = (h16)0.f;
  if (row < NB && col < NB) v = (h16)adj[((size_t)br * NB + row) * NB + col];
  adjp[i] = v;
}

// zero the padded node columns (500..511) of h_t once per call
__global__ void k_zeroht(h16* __restrict__ h_t) {
  int i = blockIdx.x * blockDim.x + threadIdx.x;
  if (i >= BB * DB * (NPAD - NB)) return;
  int node = NB + (i % (NPAD - NB));
  int bd = i / (NPAD - NB);                  // b*DB + d
  h_t[(size_t)bd * NPAD + node] = (h16)0.f;
}

// g[m, 0:512]=nc, [512:1024]=n2q, [1024:1536]=nc*n2q, [1536:2048]=nc*q2n[b]
__global__ void k_buildg(const float* __restrict__ nc, const float* __restrict__ n2q,
                         const float* __restrict__ q2n, h16* __restrict__ g) {
  int i = blockIdx.x * blockDim.x + threadIdx.x;
  if (i >= MROWS * DB) return;
  int m = i >> 9, j = i & 511;
  float a = nc[i], b2 = n2q[i], c = q2n[(m / NB) * DB + j];
  h16* gr = g + (size_t)m * 2048;
  gr[j]        = (h16)a;
  gr[512 + j]  = (h16)b2;
  gr[1024 + j] = (h16)(a * b2);
  gr[1536 + j] = (h16)(a * c);
}

// ---------------------------------------------------------------------------
// K1: nodes_compress = tanh(Ne @ Wn + bn); last0 = nc * nmask   grid (63,8)
// Each wave: 32x64 tile (2 M-subtiles x 4 N-subtiles).
// ---------------------------------------------------------------------------
__global__ __launch_bounds__(128) void k_compress(const h16* __restrict__ NeH,
                                                  const h16* __restrict__ WnT,
                                                  const float* __restrict__ bn,
                                                  const int* __restrict__ nlen,
                                                  float* __restrict__ nc,
                                                  float* __restrict__ last,
                                                  h16* __restrict__ lastH) {
  int lane = threadIdx.x & 31, wid = threadIdx.x >> 5;
  int m0 = (blockIdx.x * 4 + wid) * 32;
  int n0 = blockIdx.y * 64;
  v8f acc[2][4] = {};
  for (int k0 = 0; k0 < EB; k0 += 32) {
    v16h a0 = load_a16(NeH, EB, m0, k0, lane);
    v16h a1 = load_a16(NeH, EB, m0 + 16, k0, lane);
#pragma unroll
    for (int t = 0; t < 4; ++t) {
      v16h b = load_bt16(WnT, EB, n0 + 16 * t, k0, lane);
      acc[0][t] = wmma32(a0, b, acc[0][t]);
      acc[1][t] = wmma32(a1, b, acc[1][t]);
    }
  }
  int ncol = lane & 15;
#pragma unroll
  for (int u = 0; u < 2; ++u) {
    int mbase = m0 + 16 * u + ((lane >> 4) << 3);
#pragma unroll
    for (int t = 0; t < 4; ++t) {
      int n = n0 + 16 * t + ncol;
      float bias = bn[n];
#pragma unroll
      for (int v = 0; v < 8; ++v) {
        int m = mbase + v;
        if (m < MROWS) {
          float x = tanhf(acc[u][t][v] + bias);
          int bb = m / NB, nn = m - bb * NB;
          float mk = (nn < nlen[bb]) ? 1.f : 0.f;
          size_t o = (size_t)m * DB + n;
          nc[o] = x;
          float l = x * mk;
          last[o] = l;
          lastH[o] = (h16)l;
        }
      }
    }
  }
}

// ---------------------------------------------------------------------------
// K2: query = Qe @ Wq + bq     grid (4,8)
// ---------------------------------------------------------------------------
__global__ __launch_bounds__(128) void k_query(const h16* __restrict__ QeH,
                                               const h16* __restrict__ WqT,
                                               const float* __restrict__ bq,
                                               float* __restrict__ qry) {
  int lane = threadIdx.x & 31, wid = threadIdx.x >> 5;
  int m0 = (blockIdx.x * 4 + wid) * 32;
  if (m0 >= MQ) return;
  int n0 = blockIdx.y * 64;
  v8f acc[2][4] = {};
  for (int k0 = 0; k0 < EB; k0 += 32) {
    v16h a0 = load_a16(QeH, EB, m0, k0, lane);
    v16h a1 = load_a16(QeH, EB, m0 + 16, k0, lane);
#pragma unroll
    for (int t = 0; t < 4; ++t) {
      v16h b = load_bt16(WqT, EB, n0 + 16 * t, k0, lane);
      acc[0][t] = wmma32(a0, b, acc[0][t]);
      acc[1][t] = wmma32(a1, b, acc[1][t]);
    }
  }
  int ncol = lane & 15;
#pragma unroll
  for (int u = 0; u < 2; ++u) {
    int mbase = m0 + 16 * u + ((lane >> 4) << 3);
#pragma unroll
    for (int t = 0; t < 4; ++t) {
      int n = n0 + 16 * t + ncol;
      float bias = bq[n];
#pragma unroll
      for (int v = 0; v < 8; ++v) {
        int m = mbase + v;
        if (m < MQ) qry[(size_t)m * DB + n] = acc[u][t][v] + bias;
      }
    }
  }
}

// ---------------------------------------------------------------------------
// K3: h = (last @ Wh + bh) * nmask -> hf (f32) and h_t (f16, [b][d][node])
// grid (63,8)
// ---------------------------------------------------------------------------
__global__ __launch_bounds__(128) void k_hgemm(const h16* __restrict__ lastAH,
                                               const h16* __restrict__ WhT,
                                               const float* __restrict__ bh,
                                               const int* __restrict__ nlen,
                                               float* __restrict__ hf,
                                               h16* __restrict__ h_t) {
  int lane = threadIdx.x & 31, wid = threadIdx.x >> 5;
  int m0 = (blockIdx.x * 4 + wid) * 32;
  int n0 = blockIdx.y * 64;
  v8f acc[2][4] = {};
  for (int k0 = 0; k0 < DB; k0 += 32) {
    v16h a0 = load_a16(lastAH, DB, m0, k0, lane);
    v16h a1 = load_a16(lastAH, DB, m0 + 16, k0, lane);
#pragma unroll
    for (int t = 0; t < 4; ++t) {
      v16h b = load_bt16(WhT, DB, n0 + 16 * t, k0, lane);
      acc[0][t] = wmma32(a0, b, acc[0][t]);
      acc[1][t] = wmma32(a1, b, acc[1][t]);
    }
  }
  int ncol = lane & 15;
#pragma unroll
  for (int u = 0; u < 2; ++u) {
    int mbase = m0 + 16 * u + ((lane >> 4) << 3);
#pragma unroll
    for (int t = 0; t < 4; ++t) {
      int n = n0 + 16 * t + ncol;
      float bias = bh[n];
#pragma unroll
      for (int v = 0; v < 8; ++v) {
        int m = mbase + v;
        if (m < MROWS) {
          int bb = m / NB, nn = m - bb * NB;
          float mk = (nn < nlen[bb]) ? 1.f : 0.f;
          float x = (acc[u][t][v] + bias) * mk;
          hf[(size_t)m * DB + n] = x;
          h_t[((size_t)bb * DB + n) * NPAD + nn] = (h16)x;
        }
      }
    }
  }
}

// ---------------------------------------------------------------------------
// K4: update[b] = sum_r adj[b,r] @ h[b] + h[b]     grid (16*4, 8)
// adj padded to 512x512; B = h_t (node-major "transposed"). Branchless loop.
// B fragments hoisted across relations AND both M-subtiles: 24 wmma / k-step.
// ---------------------------------------------------------------------------
__global__ __launch_bounds__(128) void k_adjmm(const h16* __restrict__ adjp,
                                               const h16* __restrict__ h_t,
                                               const float* __restrict__ hf,
                                               float* __restrict__ upd,
                                               h16* __restrict__ updH) {
  int lane = threadIdx.x & 31, wid = threadIdx.x >> 5;
  int b = blockIdx.x >> 2;
  int m0 = ((blockIdx.x & 3) * 4 + wid) * 32;
  int n0 = blockIdx.y * 64;
  const h16* hb = h_t + (size_t)b * DB * NPAD;
  const h16* ab0 = adjp + (size_t)b * RB * NPAD * NPAD;
  v8f acc[2][4] = {};
  for (int k0 = 0; k0 < NPAD; k0 += 32) {
    v16h bf[4];
#pragma unroll
    for (int t = 0; t < 4; ++t) bf[t] = load_bt16(hb, NPAD, n0 + 16 * t, k0, lane);
#pragma unroll
    for (int r = 0; r < RB; ++r) {
      const h16* ar = ab0 + (size_t)r * NPAD * NPAD;
      v16h a0 = load_a16(ar, NPAD, m0, k0, lane);
      v16h a1 = load_a16(ar, NPAD, m0 + 16, k0, lane);
#pragma unroll
      for (int t = 0; t < 4; ++t) {
        acc[0][t] = wmma32(a0, bf[t], acc[0][t]);
        acc[1][t] = wmma32(a1, bf[t], acc[1][t]);
      }
    }
  }
  int ncol = lane & 15;
#pragma unroll
  for (int u = 0; u < 2; ++u) {
    int mbase = m0 + 16 * u + ((lane >> 4) << 3);
#pragma unroll
    for (int t = 0; t < 4; ++t) {
      int n = n0 + 16 * t + ncol;
#pragma unroll
      for (int v = 0; v < 8; ++v) {
        int m = mbase + v;
        if (m < NB) {
          size_t o = (size_t)(b * NB + m) * DB + n;
          float x = acc[u][t][v] + hf[o];
          upd[o] = x;
          updH[o] = (h16)x;
        }
      }
    }
  }
}

// ---------------------------------------------------------------------------
// K5: att = sigmoid(concat(update,last) @ Wc + bc) * nmask
//     lastB = att*tanh(update) + (1-att)*lastA     grid (63,8)
// ---------------------------------------------------------------------------
__global__ __launch_bounds__(128) void k_gate(const float* __restrict__ upd,
                                              const h16* __restrict__ updH,
                                              const float* __restrict__ lastA,
                                              const h16* __restrict__ lastAH,
                                              const h16* __restrict__ WcT,
                                              const float* __restrict__ bc,
                                              const int* __restrict__ nlen,
                                              float* __restrict__ lastB,
                                              h16* __restrict__ lastBH) {
  int lane = threadIdx.x & 31, wid = threadIdx.x >> 5;
  int m0 = (blockIdx.x * 4 + wid) * 32;
  int n0 = blockIdx.y * 64;
  v8f acc[2][4] = {};
  for (int k0 = 0; k0 < 2 * DB; k0 += 32) {
    const h16* src = (k0 < DB) ? updH : lastAH;
    int kk = (k0 < DB) ? k0 : (k0 - DB);
    v16h a0 = load_a16(src, DB, m0, kk, lane);
    v16h a1 = load_a16(src, DB, m0 + 16, kk, lane);
#pragma unroll
    for (int t = 0; t < 4; ++t) {
      v16h b = load_bt16(WcT, 2 * DB, n0 + 16 * t, k0, lane);
      acc[0][t] = wmma32(a0, b, acc[0][t]);
      acc[1][t] = wmma32(a1, b, acc[1][t]);
    }
  }
  int ncol = lane & 15;
#pragma unroll
  for (int u = 0; u < 2; ++u) {
    int mbase = m0 + 16 * u + ((lane >> 4) << 3);
#pragma unroll
    for (int t = 0; t < 4; ++t) {
      int n = n0 + 16 * t + ncol;
      float bias = bc[n];
#pragma unroll
      for (int v = 0; v < 8; ++v) {
        int m = mbase + v;
        if (m < MROWS) {
          int bb = m / NB, nn = m - bb * NB;
          float mk = (nn < nlen[bb]) ? 1.f : 0.f;
          float att = mk / (1.f + __expf(-(acc[u][t][v] + bias)));
          size_t o = (size_t)m * DB + n;
          float uu = upd[o], la = lastA[o];
          float x = att * tanhf(uu) + (1.f - att) * la;
          lastB[o] = x;
          lastBH[o] = (h16)x;
        }
      }
    }
  }
}

// ---------------------------------------------------------------------------
// K8: hidden = tanh(g @ Wo1 + bo1)   grid (63,2)
// ---------------------------------------------------------------------------
__global__ __launch_bounds__(128) void k_out1(const h16* __restrict__ gH,
                                              const h16* __restrict__ Wo1T,
                                              const float* __restrict__ bo1,
                                              float* __restrict__ hidden) {
  int lane = threadIdx.x & 31, wid = threadIdx.x >> 5;
  int m0 = (blockIdx.x * 4 + wid) * 32;
  int n0 = blockIdx.y * 64;
  v8f acc[2][4] = {};
  for (int k0 = 0; k0 < 4 * DB; k0 += 32) {
    v16h a0 = load_a16(gH, 4 * DB, m0, k0, lane);
    v16h a1 = load_a16(gH, 4 * DB, m0 + 16, k0, lane);
#pragma unroll
    for (int t = 0; t < 4; ++t) {
      v16h b = load_bt16(Wo1T, 4 * DB, n0 + 16 * t, k0, lane);
      acc[0][t] = wmma32(a0, b, acc[0][t]);
      acc[1][t] = wmma32(a1, b, acc[1][t]);
    }
  }
  int ncol = lane & 15;
#pragma unroll
  for (int u = 0; u < 2; ++u) {
    int mbase = m0 + 16 * u + ((lane >> 4) << 3);
#pragma unroll
    for (int t = 0; t < 4; ++t) {
      int n = n0 + 16 * t + ncol;
      float bias = bo1[n];
#pragma unroll
      for (int v = 0; v < 8; ++v) {
        int m = mbase + v;
        if (m < MROWS) hidden[(size_t)m * 128 + n] = tanhf(acc[u][t][v] + bias);
      }
    }
  }
}

// ---------------------------------------------------------------------------
// Attention / output VALU kernels
// ---------------------------------------------------------------------------
__global__ __launch_bounds__(128) void k_rowdots(const float* __restrict__ lastF,
                                                 const float* __restrict__ qry,
                                                 const float* __restrict__ Wa,
                                                 float* __restrict__ lastwn,
                                                 float* __restrict__ qw) {
  int lane = threadIdx.x & 31, wid = threadIdx.x >> 5;
  int row = blockIdx.x * 4 + wid;
  float acc = 0.f;
  if (row < MROWS) {
    const float* r = lastF + (size_t)row * DB;
    for (int d = lane; d < DB; d += 32) acc += r[d] * Wa[d];
  } else if (row < MROWS + MQ) {
    const float* r = qry + (size_t)(row - MROWS) * DB;
    for (int d = lane; d < DB; d += 32) acc += r[d] * Wa[DB + d];
  }
  for (int off = 16; off; off >>= 1) acc += __shfl_xor(acc, off, 32);
  if (lane == 0) {
    if (row < MROWS) lastwn[row] = acc;
    else if (row < MROWS + MQ) qw[row - MROWS] = acc;
  }
}

__global__ __launch_bounds__(32) void k_sim(const float* __restrict__ last,
                                            const float* __restrict__ qry,
                                            const float* __restrict__ Wa,
                                            const float* __restrict__ lastwn,
                                            const float* __restrict__ qw,
                                            float* __restrict__ rowmax,
                                            float* __restrict__ n2q) {
  int row = blockIdx.x;
  int lane = threadIdx.x;
  int b = row / NB;
  __shared__ float ps[32];
  float s = -__builtin_inff();
  if (lane < QB) {
    const float* lr = last + (size_t)row * DB;
    const float* qr = qry + (size_t)(b * QB + lane) * DB;
    const float* wsv = Wa + 2 * DB;
    float acc = 0.f;
    for (int d = 0; d < DB; ++d) acc += lr[d] * wsv[d] * qr[d];
    s = lastwn[row] + qw[b * QB + lane] + acc;
  }
  float mx = s;
  for (int off = 16; off; off >>= 1) mx = fmaxf(mx, __shfl_xor(mx, off, 32));
  if (lane == 0) rowmax[row] = mx;
  float p = (lane < QB) ? __expf(s - mx) : 0.f;
  float sm = p;
  for (int off = 16; off; off >>= 1) sm += __shfl_xor(sm, off, 32);
  p /= sm;
  ps[lane] = p;
  __syncthreads();
  for (int d = lane; d < DB; d += 32) {
    float acc = 0.f;
    const float* qb = qry + (size_t)(b * QB) * DB + d;
    for (int q = 0; q < QB; ++q) acc += ps[q] * qb[(size_t)q * DB];
    n2q[(size_t)row * DB + d] = acc;
  }
}

__global__ __launch_bounds__(256) void k_batt(const float* __restrict__ rowmax,
                                              const float* __restrict__ nc,
                                              float* __restrict__ q2n) {
  int b = blockIdx.x, t = threadIdx.x;
  __shared__ float red[256];
  __shared__ float batt[NB];
  const float* rm = rowmax + b * NB;
  float mx = -__builtin_inff();
  for (int n = t; n < NB; n += 256) mx = fmaxf(mx, rm[n]);
  red[t] = mx; __syncthreads();
  for (int s = 128; s; s >>= 1) { if (t < s) red[t] = fmaxf(red[t], red[t + s]); __syncthreads(); }
  mx = red[0]; __syncthreads();
  float sm = 0.f;
  for (int n = t; n < NB; n += 256) sm += __expf(rm[n] - mx);
  red[t] = sm; __syncthreads();
  for (int s = 128; s; s >>= 1) { if (t < s) red[t] += red[t + s]; __syncthreads(); }
  sm = red[0]; __syncthreads();
  for (int n = t; n < NB; n += 256) batt[n] = __expf(rm[n] - mx) / sm;
  __syncthreads();
  for (int d = t; d < DB; d += 256) {
    float acc = 0.f;
    const float* ncb = nc + (size_t)(b * NB) * DB + d;
    for (int n = 0; n < NB; ++n) acc += batt[n] * ncb[(size_t)n * DB];
    q2n[b * DB + d] = acc;
  }
}

__global__ __launch_bounds__(128) void k_raw(const float* __restrict__ hidden,
                                             const float* __restrict__ Wo2,
                                             const float* __restrict__ bo2,
                                             float* __restrict__ raw) {
  int lane = threadIdx.x & 31, wid = threadIdx.x >> 5;
  int row = blockIdx.x * 4 + wid;
  if (row >= MROWS) return;
  const float* h = hidden + (size_t)row * 128;
  float acc = 0.f;
  for (int d = lane; d < 128; d += 32) acc += h[d] * Wo2[d];
  for (int off = 16; off; off >>= 1) acc += __shfl_xor(acc, off, 32);
  if (lane == 0) raw[row] = acc + bo2[0];
}

__global__ __launch_bounds__(128) void k_out(const float* __restrict__ raw,
                                             const int* __restrict__ mask,
                                             float* __restrict__ out) {
  int lane = threadIdx.x & 31, wid = threadIdx.x >> 5;
  int idx = blockIdx.x * 4 + wid;
  if (idx >= BB * CB) return;
  int b = idx / CB;
  const int* mrow = mask + (size_t)idx * NB;
  const float* rb = raw + (size_t)b * NB;
  float mx = -__builtin_inff();
  for (int n = lane; n < NB; n += 32) {
    float p = (float)mrow[n] * rb[n];
    p = (p == 0.f) ? -__builtin_inff() : p;
    mx = fmaxf(mx, p);
  }
  for (int off = 16; off; off >>= 1) mx = fmaxf(mx, __shfl_xor(mx, off, 32));
  if (lane == 0) out[idx] = mx;
}

// ---------------------------------------------------------------------------
extern "C" void kernel_launch(void* const* d_in, const int* in_sizes, int n_in,
                              void* d_out, int out_size, void* d_ws, size_t ws_size,
                              hipStream_t stream) {
  const float* nodes_elmo = (const float*)d_in[0];
  const float* query_elmo = (const float*)d_in[1];
  const float* adj        = (const float*)d_in[2];
  const int*   mask       = (const int*)d_in[3];
  const int*   nlen       = (const int*)d_in[4];
  // d_in[5] query_length unused in the forward pass
  const float* Wq  = (const float*)d_in[6];
  const float* bq  = (const float*)d_in[7];
  const float* Wn  = (const float*)d_in[8];
  const float* bn  = (const float*)d_in[9];
  const float* Wh  = (const float*)d_in[10];
  const float* bh  = (const float*)d_in[11];
  const float* Wc  = (const float*)d_in[12];
  const float* bc  = (const float*)d_in[13];
  const float* Wa  = (const float*)d_in[14];
  const float* Wo1 = (const float*)d_in[15];
  const float* bo1 = (const float*)d_in[16];
  const float* Wo2 = (const float*)d_in[17];
  const float* bo2 = (const float*)d_in[18];
  float* out = (float*)d_out;

  char* ws = (char*)d_ws;
  size_t off = 0;
  auto alloc = [&](size_t bytes) -> void* {
    void* p = ws + off;
    off = (off + bytes + 255) & ~(size_t)255;
    return p;
  };

  const int NE_ELEMS  = MROWS * EB;             // 24,576,000
  const int QE_ELEMS  = MQ * EB;                // 1,228,800
  const int ADJP_ELEMS = BB * RB * NPAD * NPAD; // 12,582,912

  h16*   NeH    = (h16*)  alloc((size_t)NE_ELEMS * 2);
  h16*   QeH    = (h16*)  alloc((size_t)QE_ELEMS * 2);
  h16*   WnT    = (h16*)  alloc((size_t)EB * DB * 2);
  h16*   WqT    = (h16*)  alloc((size_t)EB * DB * 2);
  h16*   WhT    = (h16*)  alloc((size_t)DB * DB * 2);
  h16*   WcT    = (h16*)  alloc((size_t)2 * DB * DB * 2);
  h16*   Wo1T   = (h16*)  alloc((size_t)4 * DB * 128 * 2);
  h16*   adjp   = (h16*)  alloc((size_t)ADJP_ELEMS * 2);
  float* ncf    = (float*)alloc((size_t)MROWS * DB * 4);
  float* qryf   = (float*)alloc((size_t)MQ * DB * 4);
  float* last0  = (float*)alloc((size_t)MROWS * DB * 4);
  float* last1  = (float*)alloc((size_t)MROWS * DB * 4);
  h16*   last0H = (h16*)  alloc((size_t)MROWS * DB * 2);
  h16*   last1H = (h16*)  alloc((size_t)MROWS * DB * 2);
  float* hf     = (float*)alloc((size_t)MROWS * DB * 4);
  h16*   h_t    = (h16*)  alloc((size_t)BB * DB * NPAD * 2);
  float* upd    = (float*)alloc((size_t)MROWS * DB * 4);
  h16*   updH   = (h16*)  alloc((size_t)MROWS * DB * 2);
  float* lastwn = (float*)alloc((size_t)MROWS * 4);
  float* qw     = (float*)alloc((size_t)MQ * 4);
  float* rowmax = (float*)alloc((size_t)MROWS * 4);
  float* n2q    = (float*)alloc((size_t)MROWS * DB * 4);
  float* q2n    = (float*)alloc((size_t)BB * DB * 4);
  h16*   gH     = (h16*)  alloc((size_t)MROWS * 4 * DB * 2);
  float* hidden = (float*)alloc((size_t)MROWS * 128 * 4);
  float* raw    = (float*)alloc((size_t)MROWS * 4);
  (void)ws_size; (void)n_in; (void)in_sizes; (void)out_size;

  // ---- one-time (per call) data conditioning ----
  k_cvt_f16<<<(NE_ELEMS + 255) / 256, 256, 0, stream>>>(nodes_elmo, NeH, NE_ELEMS);
  k_cvt_f16<<<(QE_ELEMS + 255) / 256, 256, 0, stream>>>(query_elmo, QeH, QE_ELEMS);
  k_tcvt<<<(EB * DB + 255) / 256, 256, 0, stream>>>(Wn, WnT, EB, DB);
  k_tcvt<<<(EB * DB + 255) / 256, 256, 0, stream>>>(Wq, WqT, EB, DB);
  k_tcvt<<<(DB * DB + 255) / 256, 256, 0, stream>>>(Wh, WhT, DB, DB);
  k_tcvt<<<(2 * DB * DB + 255) / 256, 256, 0, stream>>>(Wc, WcT, 2 * DB, DB);
  k_tcvt<<<(4 * DB * 128 + 255) / 256, 256, 0, stream>>>(Wo1, Wo1T, 4 * DB, 128);
  k_padadj<<<(ADJP_ELEMS + 255) / 256, 256, 0, stream>>>(adj, adjp);
  k_zeroht<<<(BB * DB * (NPAD - NB) + 255) / 256, 256, 0, stream>>>(h_t);

  // ---- front-end GEMMs ----
  k_compress<<<dim3(63, 8), 128, 0, stream>>>(NeH, WnT, bn, nlen, ncf, last0, last0H);
  k_query<<<dim3(4, 8), 128, 0, stream>>>(QeH, WqT, bq, qryf);

  // ---- 5 hops (ping-pong last buffers) ----
  float* lA = last0; h16* lAH = last0H;
  float* lB = last1; h16* lBH = last1H;
  for (int hop = 0; hop < 5; ++hop) {
    k_hgemm<<<dim3(63, 8), 128, 0, stream>>>(lAH, WhT, bh, nlen, hf, h_t);
    k_adjmm<<<dim3(64, 8), 128, 0, stream>>>(adjp, h_t, hf, upd, updH);
    k_gate<<<dim3(63, 8), 128, 0, stream>>>(upd, updH, lA, lAH, WcT, bc, nlen, lB, lBH);
    float* tf = lA; lA = lB; lB = tf;
    h16* th = lAH; lAH = lBH; lBH = th;
  }

  // ---- attention block ----
  k_rowdots<<<(MROWS + MQ) / 4, 128, 0, stream>>>(lA, qryf, Wa, lastwn, qw);
  k_sim<<<MROWS, 32, 0, stream>>>(lA, qryf, Wa, lastwn, qw, rowmax, n2q);
  k_batt<<<BB, 256, 0, stream>>>(rowmax, ncf, q2n);

  // ---- output MLP ----
  k_buildg<<<(MROWS * DB + 255) / 256, 256, 0, stream>>>(ncf, n2q, q2n, gH);
  k_out1<<<dim3(63, 2), 128, 0, stream>>>(gH, Wo1T, bo1, hidden);
  k_raw<<<MROWS / 4, 128, 0, stream>>>(hidden, Wo2, bo2, raw);
  k_out<<<(BB * CB) / 4, 128, 0, stream>>>(raw, mask, out);
}
